// ButterflyLayer2D_67482526155223
// MI455X (gfx1250) — matrix operand order repaired and checked by reading, not compiled, not run on hardware
//
#include <hip/hip_runtime.h>
#include <hip/hip_bf16.h>

// CDNA5 / gfx1250 WMMA implementation of ButterflyLayer2D.
// All matmuls run on v_wmma_f32_16x16x32_f16 (wave32, f32 accumulate).
// Intermediates are f16 in d_ws (two ping-pong buffers of 128*64*64*64 halfs).
// A tiles are staged with GLOBAL_LOAD_ASYNC_TO_LDS_B128 (ASYNCcnt path,
// bypasses VGPRs); B tiles are staged pre-swizzled into fragment-major LDS so
// each lane reads one contiguous 32B run (2x ds_load_b128 per fragment).

typedef __attribute__((ext_vector_type(16))) _Float16 v16h;
typedef __attribute__((ext_vector_type(8)))  _Float16 v8h;
typedef __attribute__((ext_vector_type(8)))  float    v8f;

// Payload type of the async-to-LDS builtin (per hipcc diagnostic):
// gcc-style int vector of 4 (16 bytes).
typedef int async_b128_t __attribute__((vector_size(16)));

#define BATCH 128
#define CCH   64

// Async global->LDS 16B transfer; IOFFSET advances both addresses.
#define ASYNC_CP_B128(gsrc, ldst, off)                                       \
  __builtin_amdgcn_global_load_async_to_lds_b128(                            \
      (__attribute__((address_space(1))) async_b128_t*)(gsrc),               \
      (__attribute__((address_space(3))) async_b128_t*)(ldst), (off), 0)

__device__ __forceinline__ void wait_async_then_barrier() {
  asm volatile("s_wait_asynccnt 0x0" ::: "memory");
  __syncthreads();
}

// ---- A fragment (16x32 f16 tile, row-major LDS, ldk halfs per row) ----
// Contiguous k order (matches B swizzle): lanes 0-15 -> M=lane, K 0-15;
//             lanes 16-31 -> same M, K 16-31.
// => per lane: halfs [kh..kh+15], kh = (lane>>4)*16.
__device__ __forceinline__ v16h load_a_frag(const _Float16* A, int ldk,
                                            int row0, int k0, int lane) {
  const int row = row0 + (lane & 15);
  const int kh  = (lane >> 4) << 4;
  const _Float16* p = A + row * ldk + k0 + kh;
  const v8h lo = *(const v8h*)(p);        // K kh..kh+7
  const v8h hi = *(const v8h*)(p + 8);    // K kh+8..kh+15
  return __builtin_shufflevector(lo, hi, 0, 1, 2, 3, 4, 5, 6, 7,
                                 8, 9, 10, 11, 12, 13, 14, 15);
}

// ---- B fragment from pre-swizzled LDS: frag_idx -> 32 lanes x 16 halfs ----
__device__ __forceinline__ v16h load_b_frag(const _Float16* Bf, int frag_idx,
                                            int lane) {
  const _Float16* p = Bf + ((frag_idx << 5) + lane) * 16;
  const v8h lo = *(const v8h*)(p);
  const v8h hi = *(const v8h*)(p + 8);
  return __builtin_shufflevector(lo, hi, 0, 1, 2, 3, 4, 5, 6, 7,
                                 8, 9, 10, 11, 12, 13, 14, 15);
}

// Swizzled destination index for B element (k, n), N-tile count = ntiles.
// lane = ((k>>4)&1)*16 + (n&15), i = k&15, frag = (k>>5)*ntiles + (n>>4).
__device__ __forceinline__ int b_swz_idx(int k, int n, int ntiles) {
  const int frag = (k >> 5) * ntiles + (n >> 4);
  const int lane = (((k >> 4) & 1) << 4) + (n & 15);
  return (frag * 32 + lane) * 16 + (k & 15);
}

__device__ __forceinline__ v8f zero_v8f() {
  v8f z = {0.f, 0.f, 0.f, 0.f, 0.f, 0.f, 0.f, 0.f};
  return z;
}

// ---------------------------------------------------------------------------
// Kernel 1: patch embed.  x0[m, c] = relu(sum_k in_patch[m,k] * F[k,c] + bias)
// M = B*64*64, K = 16 (padded to 32), N = 64. 128 M-rows per block, 8 waves.
// (f32 source needs conversion -> staged through VGPRs, no async path here.)
// ---------------------------------------------------------------------------
__global__ __launch_bounds__(256) void patch_conv_kernel(
    const float* __restrict__ in_data, const float* __restrict__ in_filter,
    const float* __restrict__ in_bias, _Float16* __restrict__ x0) {
  __shared__ __attribute__((aligned(16))) _Float16 Alds[128 * 32];
  __shared__ __attribute__((aligned(16))) _Float16 Blds[32 * 64];  // swizzled
  const int tid = threadIdx.x;

  // Stage filter swizzled (16x64 real, K padded to 32 with zeros).
  for (int i = tid; i < 32 * 64; i += 256) {
    const int k = i >> 6, n = i & 63;
    const _Float16 val =
        (k < 16) ? (_Float16)in_filter[k * 64 + n] : (_Float16)0.f;
    Blds[b_swz_idx(k, n, 4)] = val;
  }
  // Stage A: 128 rows x 16 patch pixels (f32 -> f16), zero pad K 16..31.
  for (int t = tid; t < 128 * 4; t += 256) {
    const int row = t >> 2, p = t & 3;
    const int m = blockIdx.x * 128 + row;
    const int Y = m & 63, X = (m >> 6) & 63, b = m >> 12;
    const float* src = in_data + ((size_t)(b * 256 + X * 4 + p) * 256 + Y * 4);
    _Float16* dst = Alds + row * 32 + p * 4;
    dst[0] = (_Float16)src[0];
    dst[1] = (_Float16)src[1];
    dst[2] = (_Float16)src[2];
    dst[3] = (_Float16)src[3];
    if (p == 0) {
      _Float16* z = Alds + row * 32 + 16;
#pragma unroll
      for (int q = 0; q < 16; ++q) z[q] = (_Float16)0.f;
    }
  }
  __syncthreads();

  const int wave = tid >> 5, lane = tid & 31;
  const int row0 = wave * 16;
  const v16h a = load_a_frag(Alds, 32, row0, 0, lane);
#pragma unroll
  for (int nt = 0; nt < 4; ++nt) {
    v8f acc = zero_v8f();
    const v16h bf = load_b_frag(Blds, nt, lane);
    acc = __builtin_amdgcn_wmma_f32_16x16x32_f16(false, a, false, bf,
                                                 (short)0, acc, false, false);
    const int col = nt * 16 + (lane & 15);
    const float bias = in_bias[col];
#pragma unroll
    for (int g = 0; g < 8; ++g) {
      const int r = row0 + g + ((lane >> 4) << 3);
      const size_t m = (size_t)blockIdx.x * 128 + r;
      const float v = acc[g] + bias;
      x0[m * 64 + col] = (_Float16)(v > 0.f ? v : 0.f);
    }
  }
}

// ---------------------------------------------------------------------------
// Kernel 2: node conv level.  Per (u,v) node: GEMM M=B*Hout^2, K=256, N=64,
// then ReLU+bias.  Block = 128 M-rows x 64 N; K staged in 4 segments of 64
// (the 2x2 spatial taps), 2 WMMA k-steps of 32 each.  A staging is async.
// ---------------------------------------------------------------------------
__global__ __launch_bounds__(256) void node_conv_kernel(
    const _Float16* __restrict__ x_in, const float* __restrict__ w,
    const float* __restrict__ bias, _Float16* __restrict__ x_out,
    int Nu_in, int Nu_out, int Hin) {
  const int Hout = Hin >> 1;
  __shared__ __attribute__((aligned(16))) _Float16 Alds[128 * 64];  // 16 KB
  __shared__ __attribute__((aligned(16))) _Float16 Blds[256 * 64];  // 32 KB sw
  const int tid  = threadIdx.x;
  const int node = blockIdx.y;                 // u*Nu_out + v
  const int u = node / Nu_out, v = node % Nu_out;
  const int iu = (u * Nu_in) / Nu_out, iv = (v * Nu_in) / Nu_out;  // repeat map
  const float* wn = w + (size_t)node * 256 * 64;   // (x,y,c) x o, row-major
  const float* bn = bias + (size_t)node * 64;

  // Stage weights for this node: 256x64 f32 -> f16, swizzled (ntiles=4).
  for (int i = tid; i < 256 * 64; i += 256) {
    const int k = i >> 6, n = i & 63;
    Blds[b_swz_idx(k, n, 4)] = (_Float16)wn[i];
  }

  const int mbase = blockIdx.x * 128;
  const _Float16* xin_node =
      x_in + (size_t)(iu * Nu_in + iv) * (size_t)BATCH * Hin * Hin * CCH;

  const int wave = tid >> 5, lane = tid & 31;
  const int row0 = wave * 16;
  v8f acc[4];
#pragma unroll
  for (int i = 0; i < 4; ++i) acc[i] = zero_v8f();

  // Row address precompute for this thread's staging slot (64B slice).
  const int srow = tid >> 1, part = tid & 1;
  const int sm = mbase + srow;
  const int swo = sm % Hout;
  const int sho = (sm / Hout) % Hout;
  const int sb  = sm / (Hout * Hout);

#pragma unroll
  for (int seg = 0; seg < 4; ++seg) {
    const int sx = seg >> 1, sy = seg & 1;
    __syncthreads();  // all readers of Alds done (seg 0: also covers Blds)
    {
      const _Float16* src =
          xin_node + ((((size_t)sb * Hin + (2 * sho + sx)) * Hin) +
                      (2 * swo + sy)) * CCH + part * 32;
      _Float16* dst = Alds + srow * 64 + part * 32;
      // Async global->LDS: 64 bytes per thread, no VGPR round-trip.
      ASYNC_CP_B128(src, dst, 0);
      ASYNC_CP_B128(src, dst, 16);
      ASYNC_CP_B128(src, dst, 32);
      ASYNC_CP_B128(src, dst, 48);
      if (seg < 3) {  // prefetch next segment's row (global_prefetch_b8)
        const int nx = (seg + 1) >> 1, ny = (seg + 1) & 1;
        const _Float16* nsrc =
            xin_node + ((((size_t)sb * Hin + (2 * sho + nx)) * Hin) +
                        (2 * swo + ny)) * CCH + part * 32;
        __builtin_prefetch(nsrc, 0, 1);
      }
    }
    wait_async_then_barrier();
#pragma unroll
    for (int ks = 0; ks < 2; ++ks) {
      const int ktile = seg * 2 + ks;
      const v16h a = load_a_frag(Alds, 64, row0, ks * 32, lane);
#pragma unroll
      for (int nt = 0; nt < 4; ++nt) {
        const v16h bf = load_b_frag(Blds, ktile * 4 + nt, lane);
        acc[nt] = __builtin_amdgcn_wmma_f32_16x16x32_f16(
            false, a, false, bf, (short)0, acc[nt], false, false);
      }
    }
  }

  // Epilogue: ReLU + bias, store f16. Output m index == (b*Hout+ho)*Hout+wo.
  const size_t outbase = (size_t)node * BATCH * Hout * Hout * CCH;
#pragma unroll
  for (int nt = 0; nt < 4; ++nt) {
    const int col = nt * 16 + (lane & 15);
    const float bb = bn[col];
#pragma unroll
    for (int g = 0; g < 8; ++g) {
      const int r = row0 + g + ((lane >> 4) << 3);
      const size_t m = (size_t)mbase + r;
      const float val = acc[nt][g] + bb;
      x_out[outbase + m * CCH + col] = (_Float16)(val > 0.f ? val : 0.f);
    }
  }
}

// ---------------------------------------------------------------------------
// Kernel 3: head. Per node: feats(128x64) x WdT(64x128) -> out scatter.
// n = r*64 + (i*8+j); out[b][u*8+i][v*8+j][r] (f32).  A copy is async.
// ---------------------------------------------------------------------------
__global__ __launch_bounds__(256) void head_kernel(
    const _Float16* __restrict__ feats, const float* __restrict__ Wd,
    float* __restrict__ out) {
  __shared__ __attribute__((aligned(16))) _Float16 Alds[128 * 64];
  __shared__ __attribute__((aligned(16))) _Float16 Blds[64 * 128];  // swizzled
  const int tid  = threadIdx.x;
  const int node = blockIdx.x;  // u*8+v
  const int u = node >> 3, v = node & 7;

  // A: feats per node are contiguous f16 -> async 128B slice per thread.
  {
    const _Float16* fn = feats + (size_t)node * BATCH * CCH + tid * 64;
    _Float16* dst = Alds + tid * 64;
    ASYNC_CP_B128(fn, dst, 0);
    ASYNC_CP_B128(fn, dst, 16);
    ASYNC_CP_B128(fn, dst, 32);
    ASYNC_CP_B128(fn, dst, 48);
    ASYNC_CP_B128(fn, dst, 64);
    ASYNC_CP_B128(fn, dst, 80);
    ASYNC_CP_B128(fn, dst, 96);
    ASYNC_CP_B128(fn, dst, 112);
  }
  // B: K=64 (c), N=128 (n = r*64+k); swizzled (ntiles=8), f32 -> f16.
  const float* wn = Wd + (size_t)node * 2 * 64 * 64;  // (r, c, k)
  for (int i = tid; i < 64 * 128; i += 256) {
    const int c = i >> 7, n = i & 127;
    const int r = n >> 6, k = n & 63;
    Blds[b_swz_idx(c, n, 8)] = (_Float16)wn[(size_t)r * 4096 + c * 64 + k];
  }
  wait_async_then_barrier();

  const int wave = tid >> 5, lane = tid & 31;
  const int row0 = wave * 16;
#pragma unroll
  for (int nt = 0; nt < 8; ++nt) {
    v8f acc = zero_v8f();
#pragma unroll
    for (int ks = 0; ks < 2; ++ks) {
      const v16h a  = load_a_frag(Alds, 64, row0, ks * 32, lane);
      const v16h bf = load_b_frag(Blds, ks * 8 + nt, lane);
      acc = __builtin_amdgcn_wmma_f32_16x16x32_f16(false, a, false, bf,
                                                   (short)0, acc, false, false);
    }
    const int n  = nt * 16 + (lane & 15);
    const int r  = n >> 6;
    const int i8 = (n >> 3) & 7;
    const int j8 = n & 7;
#pragma unroll
    for (int g = 0; g < 8; ++g) {
      const int b = row0 + g + ((lane >> 4) << 3);
      out[(((size_t)b * 64 + (u * 8 + i8)) * 64 + (v * 8 + j8)) * 2 + r] =
          acc[g];
    }
  }
}

// ---------------------------------------------------------------------------
extern "C" void kernel_launch(void* const* d_in, const int* in_sizes, int n_in,
                              void* d_out, int out_size, void* d_ws,
                              size_t ws_size, hipStream_t stream) {
  const float* in_data   = (const float*)d_in[0];
  const float* in_filter = (const float*)d_in[1];
  const float* in_bias   = (const float*)d_in[2];
  const float* f[6];
  const float* bs[6];
  for (int l = 0; l < 6; ++l) {
    f[l]  = (const float*)d_in[3 + 2 * l];
    bs[l] = (const float*)d_in[4 + 2 * l];
  }
  const float* Wd = (const float*)d_in[15];
  float* out = (float*)d_out;

  // Two ping-pong f16 buffers of 128*64*64*64 = 33,554,432 halfs (67 MB each).
  const size_t ELEMS = (size_t)BATCH * 64 * 64 * CCH;
  _Float16* buf0 = (_Float16*)d_ws;
  _Float16* buf1 = buf0 + ELEMS;

  // Patch embed: M = 524288 rows -> 4096 blocks of 128 rows.
  patch_conv_kernel<<<4096, 256, 0, stream>>>(in_data, in_filter, in_bias,
                                              buf0);

  _Float16* cur = buf0;
  _Float16* nxt = buf1;
  int Nu_in = 1, Hin = 64;
  for (int lvl = 1; lvl <= 6; ++lvl) {
    const int Nu_out = (lvl <= 3) ? (1 << lvl) : 8;
    const int Hout   = Hin >> 1;
    const int mtiles = Hout * Hout;  // (128*Hout*Hout)/128
    dim3 grid(mtiles, Nu_out * Nu_out);
    node_conv_kernel<<<grid, 256, 0, stream>>>(cur, f[lvl - 1], bs[lvl - 1],
                                               nxt, Nu_in, Nu_out, Hin);
    _Float16* t = cur; cur = nxt; nxt = t;
    Nu_in = Nu_out;
    Hin   = Hout;
  }

  head_kernel<<<64, 256, 0, stream>>>(cur, Wd, out);
}
